// R_CLSTM_5_stereo_84885733638720
// MI455X (gfx1250) — matrix operand order repaired and checked by reading, not compiled, not run on hardware
//
#include <hip/hip_runtime.h>
#include <hip/hip_bf16.h>
#include <math.h>

// ---------------------------------------------------------------------------
// CDNA5 (gfx1250) conv-LSTM implementation.
// All convolutions are implicit GEMMs on v_wmma_f32_16x16x32_f16 (wave32).
// Activations: padded NHWC f16, C padded to 64, spatial halo (YOFF/XOFF).
// Weights: pre-packed per-lane A-fragments (16x32, contiguous k = 16*(lane/16) + j).
// Raw-input convs stage their LDS row via the Tensor Data Mover
// (tensor_load_to_lds + s_wait_tensorcnt), using D# LDS-padding for
// bank-conflict-free B-fragment reads.
// ---------------------------------------------------------------------------

typedef __attribute__((ext_vector_type(16))) _Float16 v16h;
typedef __attribute__((ext_vector_type(8)))  _Float16 h8;
typedef __attribute__((ext_vector_type(8)))  float    v8f;
typedef __attribute__((ext_vector_type(4)))  unsigned int ui32x4;
typedef __attribute__((ext_vector_type(8)))  int          i32x8;
typedef __attribute__((ext_vector_type(4)))  int          i32x4;

#if defined(__has_builtin)
#if __has_builtin(__builtin_amdgcn_tensor_load_to_lds) && \
    __has_builtin(__builtin_amdgcn_s_wait_tensorcnt)
#define HAVE_TDM 1
#endif
#endif
#ifndef HAVE_TDM
#define HAVE_TDM 0
#endif

#define HH   192
#define WW   192
#define BB   4
#define DD   8
#define HP   196            // HH + 2*YOFF
#define WP   208            // WW + 2*XOFF (8-wide halo keeps b128 alignment)
#define XOFF 8
#define YOFF 2
#define CPAD 64
#define NPIX    (BB*HH*WW)
#define NPIXPAD (BB*HP*WP)
// LDS pitches (f16 units) per x-column:
//   async/TDM path: 64 data + 4-dword pad from D# = 72  (36 dwords, 16-bank spread)
//   manual path   : 64 with XOR-swizzled 16-channel blocks
#define PITCH_ASY 72
#define PITCH_MAN 64

// ---------------------------------------------------------------------------
// Implicit-GEMM conv kernel.
//   block = 192 threads = 6 waves; one block = one output image row, one
//   16-channel output tile (blockIdx.y = mtile). Wave w computes x-columns
//   [32w, 32w+32) as two 16x16 WMMA accumulators.
//   K decomposition: (r,s) filter tap x {channels 0-31, 32-63}.
//   ASYNC: row staged by the Tensor Data Mover (raw copy; source buffer has
//          true zero halos). Otherwise manual staging with optional fused
//          BatchNorm+ReLU ('ss'), forcing out-of-image positions to zero.
// ---------------------------------------------------------------------------
template<int KS, bool ASYNC>
__launch_bounds__(192)
__global__ void conv_wmma_kernel(const _Float16* __restrict__ in,
                                 const _Float16* __restrict__ wpack,
                                 const float*    __restrict__ biasp,   // mtiles*16 or null
                                 const float2*   __restrict__ ss,      // 64 or null
                                 _Float16* __restrict__ out, int outCP)
{
    constexpr int PAD   = KS / 2;
    constexpr int NK    = KS * KS * 2;
    constexpr int PITCH = ASYNC ? PITCH_ASY : PITCH_MAN;
    constexpr int NELEM = WP * CPAD;          // f16 per staged row (13312)
    extern __shared__ char smem_raw[];        // dynamic LDS: offset 0 for TDM
    _Float16* lds = (_Float16*)smem_raw;

    const int tid  = threadIdx.x;
    const int lane = tid & 31;
    const int wv   = tid >> 5;            // 0..5
    const int mt   = blockIdx.y;
    const int bi   = blockIdx.x / HH;
    const int y    = blockIdx.x - bi * HH;
    const int n    = lane & 15;
    const int hi   = lane >> 4;           // half-wave select (K / M split)
    const int x0   = wv * 32 + n;
    const int x1   = x0 + 16;

    v8f acc0 = {};
    v8f acc1 = {};

    for (int r = 0; r < KS; ++r) {
        __syncthreads();
        const int gy = y + r - PAD;
        const _Float16* src = in + (size_t)(bi * HP + gy + YOFF) * WP * CPAD;
        bool manual = true;
        if constexpr (ASYNC) {
#if HAVE_TDM
            manual = false;
            if (wv == 0) {
                // ---- Tensor DMA: one padded NHWC row -> LDS (TENSORcnt) ----
                unsigned long long ga = (unsigned long long)(const void*)src;
                ui32x4 g0;
                g0[0] = 1u;                                   // count=1, user D#
                g0[1] = 0u;                                   // lds_addr = 0
                g0[2] = (unsigned)(ga & 0xFFFFFFFFu);         // global_addr lo
                g0[3] = (unsigned)((ga >> 32) & 0x1FFFFFFu)   // global_addr hi
                        | (2u << 30);                         // type = 2 (image)
                i32x8 g1;
                g1[0] = (1 << 16)        // data_size = 2B
                      | (1 << 20)        // pad_enable
                      | (4 << 22)        // pad_interval: every 32 dwords (128B)
                      | (3 << 25);       // pad_amount : 4 dwords (16B)
                g1[1] = (int)((NELEM & 0xFFFF) << 16);        // tensor_dim0 lo
                g1[2] = (int)(((NELEM >> 16) & 0xFFFF)        // tensor_dim0 hi
                        | (1 << 16));                         // tensor_dim1 = 1
                g1[3] = (int)((NELEM & 0xFFFF) << 16);        // tile_dim0
                g1[4] = 1;                                    // tile_dim1 = 1
                g1[5] = (int)NELEM;                           // dim0 stride lo
                g1[6] = 0;
                g1[7] = 0;
                i32x4 g2; g2[0] = 1; g2[1] = 0; g2[2] = 0; g2[3] = 0;
                i32x4 g3; g3[0] = 0; g3[1] = 0; g3[2] = 0; g3[3] = 0;
                i32x8 g4;
                g4[0] = 0; g4[1] = 0; g4[2] = 0; g4[3] = 0;
                g4[4] = 0; g4[5] = 0; g4[6] = 0; g4[7] = 0;
                // 6-arg (clang-23 / therock-10.0) form
                __builtin_amdgcn_tensor_load_to_lds(g0, g1, g2, g3, g4, 0);
                __builtin_amdgcn_s_wait_tensorcnt(0);
            }
#endif
        }
        if (manual) {
            // ---- manual staging, fused BN+ReLU, halo forced to zero ----
            const bool rowin = (gy >= 0) && (gy < HH);
            if (r + 1 < KS) {   // prefetch next filter row (global_prefetch_b8)
                const _Float16* nsrc =
                    in + (size_t)(bi * HP + y + r + 1 - PAD + YOFF) * WP * CPAD;
                __builtin_prefetch(nsrc + tid * 64, 0, 1);
            }
            for (int i = tid; i < NELEM / 8; i += 192) {
                const int xc = i >> 3;          // padded col 0..207
                const int gx = xc - XOFF;
                const int c0 = (i & 7) * 8;     // channel base 0..56
                h8 v = {};
                if (rowin && gx >= 0 && gx < WW) {
                    v = ((const h8*)src)[i];
                    if (ss) {
#pragma unroll
                        for (int j = 0; j < 8; ++j) {
                            float2 sc = ss[c0 + j];
                            float  f  = fmaxf((float)v[j] * sc.x + sc.y, 0.f);
                            v[j] = (_Float16)f;
                        }
                    }
                }
                int dst;
                if constexpr (ASYNC) {
                    dst = xc * PITCH + c0;                    // padded pitch
                } else {
                    const int blk = ((c0 >> 4) ^ xc) & 3;     // XOR swizzle
                    dst = xc * PITCH + ((c0 & 15) | (blk << 4));
                }
                *(h8*)&lds[dst] = v;
            }
        }
        __syncthreads();
        // ---- WMMA over the KS taps x 2 channel-chunks of this filter row ----
#pragma unroll
        for (int s = 0; s < KS; ++s) {
            const int xl0 = x0 + s - PAD + XOFF;
            const int xl1 = x1 + s - PAD + XOFF;
#pragma unroll
            for (int ck = 0; ck < 2; ++ck) {
                const int kstep = (r * KS + s) * 2 + ck;
                // A fragment: pre-packed, broadcast across the block
                const _Float16* ap =
                    wpack + ((size_t)(mt * NK + kstep) * 32 + lane) * 16;
                v16h a;
                ((h8*)&a)[0] = ((const h8*)ap)[0];
                ((h8*)&a)[1] = ((const h8*)ap)[1];
                // B fragments: 16 consecutive channels at pixel column n
                const int cb = ck * 32 + hi * 16;             // multiple of 16
                int o0, o1;
                if constexpr (ASYNC) {
                    o0 = xl0 * PITCH + cb;
                    o1 = xl1 * PITCH + cb;
                } else {
                    o0 = xl0 * PITCH + ((((cb >> 4) ^ xl0) & 3) << 4);
                    o1 = xl1 * PITCH + ((((cb >> 4) ^ xl1) & 3) << 4);
                }
                v16h b0, b1;
                ((h8*)&b0)[0] = *(const h8*)&lds[o0];
                ((h8*)&b0)[1] = *(const h8*)&lds[o0 + 8];
                ((h8*)&b1)[0] = *(const h8*)&lds[o1];
                ((h8*)&b1)[1] = *(const h8*)&lds[o1 + 8];
                acc0 = __builtin_amdgcn_wmma_f32_16x16x32_f16(
                           false, a, false, b0, (short)0, acc0, false, false);
                acc1 = __builtin_amdgcn_wmma_f32_16x16x32_f16(
                           false, a, false, b1, (short)0, acc1, false, false);
            }
        }
    }
    // ---- epilogue: bias, f16 convert, NHWC b128 stores ----
    const int mbase = mt * 16 + hi * 8;   // lane holds 8 consecutive M rows
    h8 e0, e1;
#pragma unroll
    for (int j = 0; j < 8; ++j) {
        float bv = biasp ? biasp[mbase + j] : 0.f;
        e0[j] = (_Float16)(acc0[j] + bv);
        e1[j] = (_Float16)(acc1[j] + bv);
    }
    const size_t rowoff = (size_t)(bi * HP + y + YOFF) * WP;
    *(h8*)&out[(rowoff + x0 + XOFF) * outCP + mbase] = e0;
    *(h8*)&out[(rowoff + x1 + XOFF) * outCP + mbase] = e1;
}

// ---------------------------------------------------------------------------
// Elementwise LSTM update: gates -> cs update -> r = [cs ; sigmoid(q) ; 0pad]
// ---------------------------------------------------------------------------
__global__ void lstm_elem_kernel(const _Float16* __restrict__ gout,
                                 float* __restrict__ cs,
                                 _Float16* __restrict__ rbuf)
{
    int p = blockIdx.x * 256 + threadIdx.x;
    if (p >= NPIX) return;
    int bi = p / (HH * WW);
    int rem = p - bi * HH * WW;
    int y = rem / WW;
    int x = rem - y * WW;
    size_t pp = (size_t)(bi * HP + y + YOFF) * WP + x + XOFF;
    const _Float16* g = gout + pp * CPAD;
    _Float16*       r = rbuf + pp * CPAD;
    float*          c = cs + (size_t)p * 8;
    h8 fz = *(const h8*)(g + 0);
    h8 iz = *(const h8*)(g + 8);
    h8 gz = *(const h8*)(g + 16);
    h8 rc;
#pragma unroll
    for (int j = 0; j < 8; ++j) {
        float f = 1.f / (1.f + __expf(-(float)fz[j]));
        float i = 1.f / (1.f + __expf(-(float)iz[j]));
        float t = tanhf((float)gz[j]);
        float nc = f * c[j] + i * t;
        c[j]  = nc;
        rc[j] = (_Float16)nc;
    }
    *(h8*)(r + 0) = rc;
#pragma unroll
    for (int k = 0; k < 4; ++k) {
        h8 q = *(const h8*)(g + 24 + k * 8);
        h8 rq;
#pragma unroll
        for (int j = 0; j < 8; ++j)
            rq[j] = (_Float16)(1.f / (1.f + __expf(-(float)q[j])));
        *(h8*)(r + 8 + k * 8) = rq;
    }
    h8 z = {};
    *(h8*)(r + 40) = z; *(h8*)(r + 48) = z; *(h8*)(r + 56) = z;
}

// ---------------------------------------------------------------------------
// Build ic = [x(32, fp32 NCHW slice -> f16) ; h(8) ; 0pad(24)] in NHWC64.
// ---------------------------------------------------------------------------
__global__ void pack_ic_kernel(const float* __restrict__ xin, int t,
                               const _Float16* __restrict__ hs,
                               _Float16* __restrict__ ic)
{
    int p = blockIdx.x * 256 + threadIdx.x;
    if (p >= NPIX) return;
    int bi = p / (HH * WW);
    int rem = p - bi * HH * WW;
    int y = rem / WW;
    int x = rem - y * WW;
    size_t pp = (size_t)(bi * HP + y + YOFF) * WP + x + XOFF;
    _Float16* o = ic + pp * CPAD;
    const float* xp = xin + (((size_t)(bi * DD + t) * 32) * HH + y) * WW + x;
#pragma unroll
    for (int c = 0; c < 32; ++c)
        o[c] = (_Float16)xp[(size_t)c * HH * WW];
    *(h8*)(o + 32) = *(const h8*)(hs + pp * 16);
    h8 z = {};
    *(h8*)(o + 40) = z; *(h8*)(o + 48) = z; *(h8*)(o + 56) = z;
}

// ---------------------------------------------------------------------------
// Training-mode BatchNorm batch statistics (sum / sumsq per channel).
// ---------------------------------------------------------------------------
__global__ void bn_stats_kernel(const _Float16* __restrict__ in,
                                float* __restrict__ st)   // [64 sums | 64 sumsq]
{
    __shared__ float red[512];
    int tid = threadIdx.x;            // 256
    int c   = tid & 63;
    int pl  = tid >> 6;               // 0..3
    float s = 0.f, q = 0.f;
    for (int p = blockIdx.x * 4 + pl; p < NPIX; p += gridDim.x * 4) {
        int bi = p / (HH * WW);
        int rem = p - bi * HH * WW;
        int y = rem / WW;
        int x = rem - y * WW;
        size_t off = ((size_t)(bi * HP + y + YOFF) * WP + x + XOFF) * CPAD + c;
        float v = (float)in[off];
        s += v; q += v * v;
    }
    red[tid] = s; red[256 + tid] = q;
    __syncthreads();
    if (tid < 64) {
        float S = red[tid] + red[tid + 64] + red[tid + 128] + red[tid + 192];
        float Q = red[256 + tid] + red[256 + tid + 64] +
                  red[256 + tid + 128] + red[256 + tid + 192];
        atomicAdd(&st[c],      S);
        atomicAdd(&st[64 + c], Q);
    }
}

__global__ void bn_finalize_kernel(const float* __restrict__ st,
                                   const float* __restrict__ gamma,
                                   const float* __restrict__ beta,
                                   int creal, float2* __restrict__ ss)
{
    int c = threadIdx.x;
    if (c >= 64) return;
    const float invn = 1.f / (float)NPIX;
    float m = st[c] * invn;
    float v = st[64 + c] * invn - m * m;
    float sc = 0.f, sh = 0.f;
    if (c < creal) {
        sc = gamma[c] * rsqrtf(v + 1e-5f);
        sh = beta[c] - m * sc;
    }
    ss[c] = make_float2(sc, sh);
}

// ---------------------------------------------------------------------------
// Scatter p_depth (NHWC16 f16, ch 0..2) to d_out fp32 (b,3,d,H,W).
// ---------------------------------------------------------------------------
__global__ void scatter_out_kernel(const _Float16* __restrict__ ot,
                                   float* __restrict__ dst, int t)
{
    int p = blockIdx.x * 256 + threadIdx.x;
    if (p >= NPIX) return;
    int bi = p / (HH * WW);
    int rem = p - bi * HH * WW;
    int y = rem / WW;
    int x = rem - y * WW;
    size_t pp = (size_t)(bi * HP + y + YOFF) * WP + x + XOFF;
    const _Float16* o = ot + pp * 16;
#pragma unroll
    for (int ch = 0; ch < 3; ++ch)
        dst[(((size_t)(bi * 3 + ch) * DD + t) * HH + y) * WW + x] = (float)o[ch];
}

// ---------------------------------------------------------------------------
// Weight packing into per-lane WMMA A-fragments (16x32, contiguous k = 16*(lane/16) + j):
//   lane<16 -> M=lane, K=kb..kb+15; lane>=16 -> kb=16.
// ---------------------------------------------------------------------------
__global__ void pack_w_kernel(const float* __restrict__ src, int Cout, int Cin,
                              int KS, int total, _Float16* __restrict__ dst)
{
    int idx = blockIdx.x * 256 + threadIdx.x;
    if (idx >= total) return;
    int NK    = KS * KS * 2;
    int j     = idx & 15;
    int lane  = (idx >> 4) & 31;
    int rest  = idx >> 9;
    int kstep = rest % NK;
    int mt    = rest / NK;
    int kb    = (lane & 16) ? 16 : 0;
    int kloc  = kb + j;
    int ck    = kstep & 1;
    int rs    = kstep >> 1;
    int r     = rs / KS;
    int s     = rs - r * KS;
    int c     = ck * 32 + kloc;
    int m     = mt * 16 + (lane & 15);
    float v = 0.f;
    if (m < Cout && c < Cin)
        v = src[(((size_t)m * Cin + c) * KS + r) * KS + s];
    dst[idx] = (_Float16)v;
}

__global__ void pack_gates_w_kernel(const float* __restrict__ Wf,
                                    const float* __restrict__ Wi,
                                    const float* __restrict__ Wc,
                                    const float* __restrict__ Wq,
                                    _Float16* __restrict__ dst)
{
    const int KS = 3, NK = 18, Cin = 40, total = 4 * NK * 512;
    int idx = blockIdx.x * 256 + threadIdx.x;
    if (idx >= total) return;
    int j     = idx & 15;
    int lane  = (idx >> 4) & 31;
    int rest  = idx >> 9;
    int kstep = rest % NK;
    int mt    = rest / NK;
    int kb    = (lane & 16) ? 16 : 0;
    int kloc  = kb + j;
    int ck    = kstep & 1;
    int rs    = kstep >> 1;
    int r     = rs / KS;
    int s     = rs - r * KS;
    int c     = ck * 32 + kloc;
    int m     = mt * 16 + (lane & 15);
    float v = 0.f;
    if (m < 56 && c < Cin) {
        const float* src; int mm;
        if      (m < 8)  { src = Wf; mm = m;      }
        else if (m < 16) { src = Wi; mm = m - 8;  }
        else if (m < 24) { src = Wc; mm = m - 16; }
        else             { src = Wq; mm = m - 24; }
        v = src[(((size_t)mm * Cin + c) * KS + r) * KS + s];
    }
    dst[idx] = (_Float16)v;
}

__global__ void pack_bias_kernel(const float* __restrict__ src, int n,
                                 float* __restrict__ dst, int ntot)
{
    int i = threadIdx.x;
    if (i < ntot) dst[i] = (i < n) ? src[i] : 0.f;
}

__global__ void pack_gate_bias_kernel(const float* bf, const float* bi,
                                      const float* bc, const float* bq,
                                      float* dst)
{
    int i = threadIdx.x;
    if (i >= 64) return;
    float v = 0.f;
    if      (i < 8)  v = bf[i];
    else if (i < 16) v = bi[i - 8];
    else if (i < 24) v = bc[i - 16];
    else if (i < 56) v = bq[i - 24];
    dst[i] = v;
}

__global__ void zero_ws_kernel(uint4* p, long n16)
{
    long i = (long)blockIdx.x * blockDim.x + threadIdx.x;
    for (; i < n16; i += (long)gridDim.x * blockDim.x)
        p[i] = make_uint4(0, 0, 0, 0);
}

__global__ void zero_f32_kernel(float* p, int n)
{
    int i = threadIdx.x;
    if (i < n) p[i] = 0.f;
}

// ---------------------------------------------------------------------------
extern "C" void kernel_launch(void* const* d_in, const int* in_sizes, int n_in,
                              void* d_out, int out_size, void* d_ws, size_t ws_size,
                              hipStream_t stream)
{
    const float* in1 = (const float*)d_in[0];
    const float* in2 = (const float*)d_in[1];
    const float* Wf  = (const float*)d_in[2];  const float* bf  = (const float*)d_in[3];
    const float* Wi  = (const float*)d_in[4];  const float* bi  = (const float*)d_in[5];
    const float* Wc  = (const float*)d_in[6];  const float* bc  = (const float*)d_in[7];
    const float* Wq  = (const float*)d_in[8];  const float* bq  = (const float*)d_in[9];
    const float* W0  = (const float*)d_in[10]; const float* g0  = (const float*)d_in[11];
    const float* be0 = (const float*)d_in[12];
    const float* W1  = (const float*)d_in[13]; const float* g1  = (const float*)d_in[14];
    const float* be1 = (const float*)d_in[15];
    const float* W2  = (const float*)d_in[16]; const float* b2w = (const float*)d_in[17];
    const float* Wh  = (const float*)d_in[18]; const float* bh  = (const float*)d_in[19];

    // ---- workspace carve-out (256B aligned) ----
    char* ws = (char*)d_ws;
    size_t off = 0;
    auto alloc = [&](size_t bytes) -> char* {
        char* p = ws + off;
        off = (off + bytes + 255) & ~(size_t)255;
        return p;
    };
    const size_t BUF64 = (size_t)NPIXPAD * 64 * 2;
    const size_t BUF16 = (size_t)NPIXPAD * 16 * 2;
    _Float16* icb  = (_Float16*)alloc(BUF64);
    _Float16* gob  = (_Float16*)alloc(BUF64);
    _Float16* rb   = (_Float16*)alloc(BUF64);
    _Float16* y0b  = (_Float16*)alloc(BUF64);
    _Float16* y1b  = (_Float16*)alloc(BUF64);
    _Float16* hsb  = (_Float16*)alloc(BUF16);
    _Float16* otb  = (_Float16*)alloc(BUF16);
    float*    csb  = (float*)alloc((size_t)NPIX * 8 * 4);
    float*    st0  = (float*)alloc(128 * 4);
    float2*   ss0  = (float2*)alloc(64 * 8);
    float*    st1  = (float*)alloc(128 * 4);
    float2*   ss1  = (float2*)alloc(64 * 8);
    _Float16* pwG  = (_Float16*)alloc((size_t)4 * 18 * 512 * 2);
    _Float16* pw0  = (_Float16*)alloc((size_t)3 * 50 * 512 * 2);
    _Float16* pw1  = (_Float16*)alloc((size_t)3 * 50 * 512 * 2);
    _Float16* pwh  = (_Float16*)alloc((size_t)1 * 18 * 512 * 2);
    _Float16* pw2  = (_Float16*)alloc((size_t)1 * 50 * 512 * 2);
    float*    bG   = (float*)alloc(64 * 4);
    float*    bH   = (float*)alloc(16 * 4);
    float*    bO   = (float*)alloc(16 * 4);
    const size_t used = off;

    // zero workspace (initial h/c state, clean halos/stats) — deterministic
    zero_ws_kernel<<<2048, 256, 0, stream>>>((uint4*)ws, (long)(used / 16));

    // weight / bias packing (once per launch)
    pack_gates_w_kernel<<<(4 * 18 * 512 + 255) / 256, 256, 0, stream>>>(Wf, Wi, Wc, Wq, pwG);
    pack_w_kernel<<<(3 * 50 * 512 + 255) / 256, 256, 0, stream>>>(W0, 40, 40, 5, 3 * 50 * 512, pw0);
    pack_w_kernel<<<(3 * 50 * 512 + 255) / 256, 256, 0, stream>>>(W1, 40, 40, 5, 3 * 50 * 512, pw1);
    pack_w_kernel<<<(1 * 18 * 512 + 255) / 256, 256, 0, stream>>>(Wh, 8, 40, 3, 1 * 18 * 512, pwh);
    pack_w_kernel<<<(1 * 50 * 512 + 255) / 256, 256, 0, stream>>>(W2, 3, 40, 5, 1 * 50 * 512, pw2);
    pack_gate_bias_kernel<<<1, 64, 0, stream>>>(bf, bi, bc, bq, bG);
    pack_bias_kernel<<<1, 16, 0, stream>>>(bh, 8, bH, 16);
    pack_bias_kernel<<<1, 16, 0, stream>>>(b2w, 3, bO, 16);

    float* outp = (float*)d_out;
    const size_t OHALF = (size_t)BB * 3 * DD * HH * WW;
    const int NB = (NPIX + 255) / 256;
    const unsigned SH_ASY = WP * PITCH_ASY * 2;   // 29952 B (TDM-padded pitch)
    const unsigned SH_MAN = WP * PITCH_MAN * 2;   // 26624 B (swizzled pitch)
    dim3 cblk(192);

    for (int t = 0; t < DD; ++t) {
        for (int half = 0; half < 2; ++half) {
            const float* xin = half ? in2 : in1;
            pack_ic_kernel<<<NB, 256, 0, stream>>>(xin, t, hsb, icb);
            // gate conv 3x3 40->56 (pad 64), TDM-staged rows
            conv_wmma_kernel<3, true><<<dim3(BB * HH, 4), cblk, SH_ASY, stream>>>(
                icb, pwG, bG, nullptr, gob, 64);
            lstm_elem_kernel<<<NB, 256, 0, stream>>>(gob, csb, rb);
            // refine conv0 5x5 40->40 (raw pre-BN output), TDM-staged rows
            conv_wmma_kernel<5, true><<<dim3(BB * HH, 3), cblk, SH_ASY, stream>>>(
                rb, pw0, nullptr, nullptr, y0b, 64);
            zero_f32_kernel<<<1, 128, 0, stream>>>(st0, 128);
            bn_stats_kernel<<<256, 256, 0, stream>>>(y0b, st0);
            bn_finalize_kernel<<<1, 64, 0, stream>>>(st0, g0, be0, 40, ss0);
            // refine conv1 5x5 40->40, BN0+ReLU fused into staging
            conv_wmma_kernel<5, false><<<dim3(BB * HH, 3), cblk, SH_MAN, stream>>>(
                y0b, pw1, nullptr, ss0, y1b, 64);
            zero_f32_kernel<<<1, 128, 0, stream>>>(st1, 128);
            bn_stats_kernel<<<256, 256, 0, stream>>>(y1b, st1);
            bn_finalize_kernel<<<1, 64, 0, stream>>>(st1, g1, be1, 40, ss1);
            // h-state conv 3x3 40->8 and depth head 5x5 40->3, BN1+ReLU fused
            conv_wmma_kernel<3, false><<<dim3(BB * HH, 1), cblk, SH_MAN, stream>>>(
                y1b, pwh, bH, ss1, hsb, 16);
            conv_wmma_kernel<5, false><<<dim3(BB * HH, 1), cblk, SH_MAN, stream>>>(
                y1b, pw2, bO, ss1, otb, 16);
            scatter_out_kernel<<<NB, 256, 0, stream>>>(otb, outp + half * OHALF, t);
        }
    }
}
